// CroppingGraph_49065706390290
// MI455X (gfx1250) — matrix operand order repaired and checked by reading, not compiled, not run on hardware
//
#include <hip/hip_runtime.h>
#include <hip/hip_bf16.h>
#include <cstdint>
#include <cstddef>

// ---------------------------------------------------------------------------
// CDNA5 (gfx1250) fused graph kernel.
// adj = f(xm @ xn^T), xr = adj @ (x@Wr), gating, score — all GEMMs on
// v_wmma_f32_16x16x32_f16 (wave32).
// Round 2: 64-row M-tiles in the N×N pass (4x less L2 re-read traffic),
// 4 n-tiles per A-fragment in the projection GEMM.
// ---------------------------------------------------------------------------

#define NROWS 8192
#define DIN   512
#define DOUT  256

typedef _Float16 HT;
typedef __attribute__((ext_vector_type(16))) _Float16 v16h;
typedef __attribute__((ext_vector_type(8)))  _Float16 v8h;
typedef __attribute__((ext_vector_type(8)))  float    v8f;

union AFrag { v16h v; v8h h[2]; };

// A fragment (16x32 f16, MxK): lane l holds row (l&15); e[0..7] = K chunk at
// kc + ((l<16)?0:8); e[8..15] = K chunk at kc+16 + ((l<16)?0:8).
__device__ __forceinline__ v16h load_a16(const HT* base, int row_stride,
                                         int row, int kc, int half_id) {
    const HT* r = base + (size_t)row * row_stride + kc + half_id * 8;
    AFrag f;
    f.h[0] = *reinterpret_cast<const v8h*>(r);
    f.h[1] = *reinterpret_cast<const v8h*>(r + 16);
    return f.v;
}

// B fragment (32x16 f16, KxN) from an N-major (transposed) array:
// lane l holds column (l&15); e[0..7] = K chunk at kc + ((l<16)?0:8);
// e[8..15] = K chunk at kc+16 + ((l<16)?0:8).
__device__ __forceinline__ v16h load_b16(const HT* base, int col_stride,
                                         int col, int kc, int half_id) {
    const HT* r = base + (size_t)col * col_stride + kc + half_id * 8;
    AFrag f;
    f.h[0] = *reinterpret_cast<const v8h*>(r);
    f.h[1] = *reinterpret_cast<const v8h*>(r + 16);
    return f.v;
}

__device__ __forceinline__ v8f wmma_f16(v16h a, v16h b, v8f c) {
    return __builtin_amdgcn_wmma_f32_16x16x32_f16(
        /*neg_a=*/false, a, /*neg_b=*/false, b,
        /*c_mod=*/(short)0, c, /*reuse_a=*/false, /*reuse_b=*/false);
}

// ---------------------------------------------------------------------------
// Elementwise f32 -> f16 copy
__global__ __launch_bounds__(256) void k_cvt(const float* __restrict__ in,
                                             HT* __restrict__ out, int n) {
    int t = blockIdx.x * 256 + threadIdx.x;
    if (t < n) out[t] = (HT)in[t];
}

// Transpose + convert: in[K,Nc] f32 -> out[Nc,K] f16
__global__ __launch_bounds__(256) void k_tr(const float* __restrict__ in,
                                            HT* __restrict__ out, int K, int Nc) {
    int t = blockIdx.x * 256 + threadIdx.x;
    if (t < K * Nc) {
        int k = t / Nc, n = t % Nc;          // coalesced read of `in`
        out[(size_t)n * K + k] = (HT)in[(size_t)k * Nc + n];
    }
}

// ---------------------------------------------------------------------------
// Four input projections: Y = xh[8192,512] @ W[512,256], f16 WMMA.
// Each wave owns a 16x64 output strip (4 n-tiles) so every A fragment is
// reused 4x. blockIdx.y selects the matrix; outputs stored in the layout the
// later kernels want (xwr transposed for contiguous B fragments).
__global__ __launch_bounds__(256) void k_proj(
    const HT* __restrict__ xh,
    const HT* __restrict__ WmT, const HT* __restrict__ WnT,
    const HT* __restrict__ WrT, const HT* __restrict__ WtT,
    HT* __restrict__ xmh, HT* __restrict__ xnh,
    HT* __restrict__ xwrT, HT* __restrict__ xlh,
    float* __restrict__ xl32) {
    const int lane = threadIdx.x & 31;
    const int wave = threadIdx.x >> 5;
    const int colL = lane & 15;
    const int h    = lane >> 4;
    const int rb   = h * 8;
    const int mat  = blockIdx.y;
    const int task = blockIdx.x * 8 + wave;      // 2048 tasks (512 m x 4 ng)
    const int m0   = (task >> 2) * 16;
    const int nb   = (task & 3) * 64;            // 4 n-tiles per wave

    const HT* WT = (mat == 0) ? WmT : (mat == 1) ? WnT : (mat == 2) ? WrT : WtT;

    v8f c[4] = {{}, {}, {}, {}};
    for (int kc = 0; kc < DIN; kc += 32) {
        v16h a = load_a16(xh, DIN, m0 + colL, kc, h);
        for (int t = 0; t < 4; ++t) {
            v16h b = load_b16(WT, DIN, nb + t * 16 + colL, kc, h);
            c[t] = wmma_f16(a, b, c[t]);
        }
    }

    for (int t = 0; t < 4; ++t) {
        const int n0 = nb + t * 16;
        if (mat == 0 || mat == 1) {
            HT* dst = (mat == 0) ? xmh : xnh;
            for (int v = 0; v < 8; ++v)
                dst[(size_t)(m0 + rb + v) * DOUT + n0 + colL] = (HT)c[t][v];
        } else if (mat == 2) {
            // store transposed [256, 8192]: per lane 8 contiguous rows -> b128
            v8h hv;
            for (int v = 0; v < 8; ++v) hv[v] = (HT)c[t][v];
            *reinterpret_cast<v8h*>(xwrT + (size_t)(n0 + colL) * NROWS + m0 + rb) = hv;
        } else {
            for (int v = 0; v < 8; ++v) {
                size_t o = (size_t)(m0 + rb + v) * DOUT + n0 + colL;
                xl32[o] = c[t][v];
                xlh[o]  = (HT)c[t][v];
            }
        }
    }
}

// ---------------------------------------------------------------------------
// Row sum-of-squares for xm (y==0) and xn (y==1). One wave per row.
__global__ __launch_bounds__(256) void k_norms(const HT* __restrict__ xmh,
                                               const HT* __restrict__ xnh,
                                               float* __restrict__ nm,
                                               float* __restrict__ nn) {
    const int lane = threadIdx.x & 31;
    const int wave = threadIdx.x >> 5;
    const int row  = blockIdx.x * 8 + wave;
    const HT* src  = (blockIdx.y == 0) ? xmh : xnh;
    float s = 0.f;
    for (int i = 0; i < DOUT; i += 32) {
        float v = (float)src[(size_t)row * DOUT + i + lane];
        s += v * v;
    }
    for (int off = 16; off >= 1; off >>= 1) s += __shfl_xor(s, off, 32);
    if (lane == 0) ((blockIdx.y == 0) ? nm : nn)[row] = s;
}

// ---------------------------------------------------------------------------
// Fused: adj tile (WMMA S + elementwise) -> d_out, and xr += adj @ xwr (WMMA).
// Block = 8 waves = 4 row-strips x 2 j-subtiles, covering a 64-row M-tile
// (4x fewer cross-block re-reads of xn/xwr than a 16-row tile). Per j-chunk
// of 32 columns: wave (ms,js) computes the S tile at (strip ms, subtile js),
// stages f16 adj in its strip's LDS buffer, then consumes the strip's full
// 16x32 adj block as a K=32 A-fragment against 8 n-tiles of xwrT
// (js=0 -> n 0..127, js=1 -> n 128..255).
__global__ __launch_bounds__(256) void k_adj(
    const HT* __restrict__ xmh, const HT* __restrict__ xnh,
    const HT* __restrict__ xwrT,
    const float* __restrict__ nm, const float* __restrict__ nn,
    float* __restrict__ adj, float* __restrict__ xr32, HT* __restrict__ xrh) {
    __shared__ __align__(16) HT lds[4 * 16 * 40];   // 4 strips x 16 rows x (32+8 pad)

    const int lane = threadIdx.x & 31;
    const int wave = threadIdx.x >> 5;
    const int ms   = wave >> 1;        // row strip 0..3
    const int js   = wave & 1;         // j sub-tile 0..1
    const int colL = lane & 15;
    const int h    = lane >> 4;
    const int rb   = h * 8;
    const int m0   = blockIdx.x * 64 + ms * 16;

    // invariant A fragments of xm for this 16-row strip (K = 256)
    v16h am[8];
    for (int kf = 0; kf < 8; ++kf)
        am[kf] = load_a16(xmh, DOUT, m0 + colL, kf * 32, h);
    float nmv[8];
    for (int v = 0; v < 8; ++v) nmv[v] = nm[m0 + rb + v];

    HT* strip = lds + ms * 16 * 40;
    v8f acc[8] = {{}, {}, {}, {}, {}, {}, {}, {}};

    for (int jc = 0; jc < NROWS; jc += 32) {
        const int j0 = jc + js * 16;

        // S = xm @ xn^T  (16x16 tile, K=256)
        v8f s = {};
        for (int kf = 0; kf < 8; ++kf) {
            v16h b = load_b16(xnh, DOUT, j0 + colL, kf * 32, h);
            s = wmma_f16(am[kf], b, s);
        }
        __builtin_prefetch(xnh + (size_t)(j0 + 32 + colL) * DOUT, 0, 1);

        const float nnj = nn[j0 + colL];
        for (int v = 0; v < 8; ++v) {
            const int grow = m0 + rb + v;
            const int gcol = j0 + colL;
            float sq = nmv[v] + nnj - 2.0f * s[v];
            float d  = sqrtf(fmaxf(sq, 0.0f)) * 0.5f;
            float e  = __expf(-d) * 0.0078125f;      // / (N/64)
            float av = (grow == gcol) ? 1.0f : e;
            adj[(size_t)grow * NROWS + gcol] = av;
            strip[(rb + v) * 40 + js * 16 + colL] = (HT)av;
        }
        __syncthreads();

        // xr_acc += adj(16x32 of this strip) @ xwr(32 x 128-per-wave)
        const HT* r = strip + colL * 40 + h * 8;
        AFrag f;
        f.h[0] = *reinterpret_cast<const v8h*>(r);
        f.h[1] = *reinterpret_cast<const v8h*>(r + 16);
        for (int t = 0; t < 8; ++t) {
            const int nt = js * 8 + t;
            v16h b = load_b16(xwrT, NROWS, nt * 16 + colL, jc, h);
            acc[t] = wmma_f16(f.v, b, acc[t]);
        }
        __syncthreads();
    }

    for (int t = 0; t < 8; ++t) {
        const int n0 = (js * 8 + t) * 16 + colL;
        for (int v = 0; v < 8; ++v) {
            const int grow = m0 + rb + v;
            size_t o = (size_t)grow * DOUT + n0;
            xr32[o] = acc[t][v];
            xrh[o]  = (HT)acc[t][v];
        }
    }
}

// ---------------------------------------------------------------------------
// Gating + score: w = sigmoid(xr@Wrg + brg + xl@Wlg + blg);
// feat = (1-w)xr + w*xl; score = feat @ Wp + bp.  WMMA for both gate GEMMs,
// shuffle-reduce across the 16-lane column group for the final dot.
__global__ __launch_bounds__(256) void k_gate(
    const HT* __restrict__ xrh, const HT* __restrict__ xlh,
    const float* __restrict__ xr32, const float* __restrict__ xl32,
    const HT* __restrict__ WrgT, const HT* __restrict__ WlgT,
    const float* __restrict__ brg, const float* __restrict__ blg,
    const float* __restrict__ Wp, const float* __restrict__ bp,
    float* __restrict__ score) {
    const int lane = threadIdx.x & 31;
    const int wave = threadIdx.x >> 5;
    const int colL = lane & 15;
    const int h    = lane >> 4;
    const int rb   = h * 8;
    const int m0   = blockIdx.x * 128 + wave * 16;

    v16h ar[8], al[8];
    for (int kf = 0; kf < 8; ++kf) {
        ar[kf] = load_a16(xrh, DOUT, m0 + colL, kf * 32, h);
        al[kf] = load_a16(xlh, DOUT, m0 + colL, kf * 32, h);
    }

    float sacc[8] = {0.f, 0.f, 0.f, 0.f, 0.f, 0.f, 0.f, 0.f};
    const float bpv = bp[0];

    for (int nt = 0; nt < 16; ++nt) {
        v8f g1 = {}, g2 = {};
        for (int kf = 0; kf < 8; ++kf) {
            v16h b1 = load_b16(WrgT, DOUT, nt * 16 + colL, kf * 32, h);
            g1 = wmma_f16(ar[kf], b1, g1);
        }
        for (int kf = 0; kf < 8; ++kf) {
            v16h b2 = load_b16(WlgT, DOUT, nt * 16 + colL, kf * 32, h);
            g2 = wmma_f16(al[kf], b2, g2);
        }
        const int n = nt * 16 + colL;
        const float b1s = brg[n], b2s = blg[n], wpn = Wp[n];
        for (int v = 0; v < 8; ++v) {
            const int grow = m0 + rb + v;
            float z  = g1[v] + b1s + g2[v] + b2s;
            float wv = 1.0f / (1.0f + __expf(-z));
            float xrv = xr32[(size_t)grow * DOUT + n];
            float xlv = xl32[(size_t)grow * DOUT + n];
            float feat = (1.0f - wv) * xrv + wv * xlv;
            sacc[v] += feat * wpn;
        }
    }

    for (int v = 0; v < 8; ++v) {
        float val = sacc[v];
        for (int off = 8; off >= 1; off >>= 1) val += __shfl_xor(val, off, 16);
        if (colL == 0) score[m0 + rb + v] = val + bpv;
    }
}

// ---------------------------------------------------------------------------
extern "C" void kernel_launch(void* const* d_in, const int* in_sizes, int n_in,
                              void* d_out, int out_size, void* d_ws, size_t ws_size,
                              hipStream_t stream) {
    const float* x   = (const float*)d_in[0];
    const float* Wm  = (const float*)d_in[1];
    const float* Wn  = (const float*)d_in[2];
    const float* Wr  = (const float*)d_in[3];
    const float* Wt  = (const float*)d_in[4];
    const float* Wrg = (const float*)d_in[5];
    const float* brg = (const float*)d_in[6];
    const float* Wlg = (const float*)d_in[7];
    const float* blg = (const float*)d_in[8];
    const float* Wp  = (const float*)d_in[9];
    const float* bp  = (const float*)d_in[10];

    char* ws = (char*)d_ws;
    auto alloc = [&](size_t bytes) -> void* {
        void* p = (void*)ws;
        ws += (bytes + 255) & ~(size_t)255;
        return p;
    };
    HT* xh    = (HT*)alloc((size_t)NROWS * DIN * 2);
    HT* WmT   = (HT*)alloc((size_t)DOUT * DIN * 2);
    HT* WnT   = (HT*)alloc((size_t)DOUT * DIN * 2);
    HT* WrT   = (HT*)alloc((size_t)DOUT * DIN * 2);
    HT* WtT   = (HT*)alloc((size_t)DOUT * DIN * 2);
    HT* WrgT  = (HT*)alloc((size_t)DOUT * DOUT * 2);
    HT* WlgT  = (HT*)alloc((size_t)DOUT * DOUT * 2);
    HT* xmh   = (HT*)alloc((size_t)NROWS * DOUT * 2);
    HT* xnh   = (HT*)alloc((size_t)NROWS * DOUT * 2);
    HT* xwrT  = (HT*)alloc((size_t)DOUT * NROWS * 2);
    HT* xlh   = (HT*)alloc((size_t)NROWS * DOUT * 2);
    float* xl32 = (float*)alloc((size_t)NROWS * DOUT * 4);
    float* xr32 = (float*)alloc((size_t)NROWS * DOUT * 4);
    HT* xrh   = (HT*)alloc((size_t)NROWS * DOUT * 2);
    float* nm = (float*)alloc((size_t)NROWS * 4);
    float* nn = (float*)alloc((size_t)NROWS * 4);

    float* adj   = (float*)d_out;
    float* score = adj + (size_t)NROWS * NROWS;

    // 1) precision / layout prep
    k_cvt<<<(NROWS * DIN + 255) / 256, 256, 0, stream>>>(x, xh, NROWS * DIN);
    k_tr<<<(DIN * DOUT + 255) / 256, 256, 0, stream>>>(Wm, WmT, DIN, DOUT);
    k_tr<<<(DIN * DOUT + 255) / 256, 256, 0, stream>>>(Wn, WnT, DIN, DOUT);
    k_tr<<<(DIN * DOUT + 255) / 256, 256, 0, stream>>>(Wr, WrT, DIN, DOUT);
    k_tr<<<(DIN * DOUT + 255) / 256, 256, 0, stream>>>(Wt, WtT, DIN, DOUT);
    k_tr<<<(DOUT * DOUT + 255) / 256, 256, 0, stream>>>(Wrg, WrgT, DOUT, DOUT);
    k_tr<<<(DOUT * DOUT + 255) / 256, 256, 0, stream>>>(Wlg, WlgT, DOUT, DOUT);

    // 2) four projections (WMMA, 4 n-tiles per A fragment)
    k_proj<<<dim3(256, 4), 256, 0, stream>>>(xh, WmT, WnT, WrT, WtT,
                                             xmh, xnh, xwrT, xlh, xl32);

    // 3) row norms for the distance trick
    k_norms<<<dim3(NROWS / 8, 2), 256, 0, stream>>>(xmh, xnh, nm, nn);

    // 4) fused adj + graph conv (WMMA, 64-row M-tiles)
    k_adj<<<NROWS / 64, 256, 0, stream>>>(xmh, xnh, xwrT, nm, nn,
                                          adj, xr32, xrh);

    // 5) gating + score (WMMA + shuffle reduce)
    k_gate<<<NROWS / 128, 256, 0, stream>>>(xrh, xlh, xr32, xl32, WrgT, WlgT,
                                            brg, blg, Wp, bp, score);
}
